// CriticAttention_35218731827275
// MI455X (gfx1250) — matrix operand order repaired and checked by reading, not compiled, not run on hardware
//
#include <hip/hip_runtime.h>
#include <hip/hip_bf16.h>

typedef _Float16 h16;
typedef __attribute__((ext_vector_type(16))) _Float16 v16h;
typedef __attribute__((ext_vector_type(8)))  _Float16 v8h;
typedef __attribute__((ext_vector_type(8)))  float    v8f;
typedef __attribute__((ext_vector_type(4)))  float    v4f;

#define D_    256
#define HID_  512
#define H_    8
#define HD_   64
#define A_    16
#define B_    8192
#define MB_   16
#define ROWS_ 256
#define OTH_  240

// d_ws layout (bytes)
#define WS_STATS 0                      // 1024 f32: sum_all, ss_all, sum_cur, ss_cur
#define WS_NORM  4096                   // 1024 f32: mean_cur, rstd_cur, mean_oth, rstd_oth
#define WS_BS2   8192                   // 512 f32 folded bias (Ws path)
#define WS_BC2   10240                  // 512 f32 folded bias (Wc path)
#define WS_WST   12288                  // f16 [512][256]  rstd_cur-scaled Ws, transposed
#define WS_WCT   (WS_WST + 262144)      // f16 [512][256]
#define WS_WQT   (WS_WCT + 262144)      // f16 [512][512]  Wq transposed: [h*64+d][n]
#define WS_WKT   (WS_WQT + 524288)
#define WS_WVT   (WS_WKT + 524288)

#define WMMA(a, b, c) __builtin_amdgcn_wmma_f32_16x16x32_f16( \
    false, (a), false, (b), (short)0, (c), false, false)

union AF { v16h v; v8h h[2]; };

// A-fragment (16x32 f16) from LDS, row-major with given stride (halves).
__device__ __forceinline__ v16h ldA(const h16* base, int stride, int row, int k0, int lane) {
    const h16* p = base + row * stride + k0 + (lane & 16);
    AF f;
    f.h[0] = *(const v8h*)p;
    f.h[1] = *(const v8h*)(p + 8);
    return f.v;
}

// B-fragment (32x16 f16) from global pre-transposed weights W[c][n] (stride halves).
__device__ __forceinline__ v16h gB(const h16* W, int stride, int cbase, int k0, int lane) {
    const h16* p = W + (size_t)(cbase + (lane & 15)) * stride + k0 + (lane & 16);
    return *(const v16h*)p;
}

// B-fragment from LDS staging buffer (padded stride 72, 2x16B loads).
__device__ __forceinline__ v16h ldB(const h16* buf, int nbase, int k0, int lane) {
    const h16* p = buf + (nbase + (lane & 15)) * 72 + k0 + (lane & 16);
    AF f;
    f.h[0] = *(const v8h*)p;
    f.h[1] = *(const v8h*)(p + 8);
    return f.v;
}

// Async DMA 16 bytes global -> LDS (no VGPR data), tracked by ASYNCcnt.
__device__ __forceinline__ void async_cp16(const h16* gsrc, h16* ldst) {
    unsigned l = (unsigned)(uintptr_t)ldst;   // low 32 bits of generic ptr = LDS offset
    asm volatile("global_load_async_to_lds_b128 %0, %1, off"
                 :: "v"(l), "v"(gsrc) : "memory");
}
__device__ __forceinline__ void wait_async0() {
    asm volatile("s_wait_asynccnt 0x0" ::: "memory");
}

// ---------------- prep kernels ----------------

__global__ void ca_zero(float* p) { p[blockIdx.x * 256 + threadIdx.x] = 0.f; }

__global__ void ca_stats(const float* __restrict__ state, const int* __restrict__ aiP,
                         float* __restrict__ stats) {
    const int f = threadIdx.x;                 // feature 0..255
    const int ai = aiP[0];
    const size_t row0 = (size_t)blockIdx.x * 256;
    float sa = 0.f, qa = 0.f, sc = 0.f, qc = 0.f;
    for (int r = 0; r < 256; ++r) {
        size_t row = row0 + r;
        float x = state[row * D_ + f];
        sa += x; qa += x * x;
        if ((int)(row & (A_ - 1)) == ai) { sc += x; qc += x * x; }
    }
    atomicAdd(&stats[f], sa);
    atomicAdd(&stats[256 + f], qa);
    atomicAdd(&stats[512 + f], sc);
    atomicAdd(&stats[768 + f], qc);
}

__global__ void ca_finalize(const float* __restrict__ stats, float* __restrict__ norm) {
    int n = threadIdx.x;
    float sa = stats[n], qa = stats[256 + n], sc = stats[512 + n], qc = stats[768 + n];
    float mc = sc / (float)B_;
    float vc = qc / (float)B_ - mc * mc;
    float so = sa - sc, qo = qa - qc;
    const float cnt = (float)B_ * 15.f;
    float mo = so / cnt;
    float vo = qo / cnt - mo * mo;
    norm[n]       = mc;
    norm[256 + n] = rsqrtf(vc + 1e-3f);
    norm[512 + n] = mo;
    norm[768 + n] = rsqrtf(vo + 1e-3f);
}

__global__ void ca_prep(const float* __restrict__ Ws, const float* __restrict__ bs,
                        const float* __restrict__ Wc, const float* __restrict__ bc,
                        const float* __restrict__ Wk, const float* __restrict__ Wq,
                        const float* __restrict__ Wv, const float* __restrict__ norm,
                        char* __restrict__ ws) {
    __shared__ float red[256];
    const int c = blockIdx.x;    // output col 0..511
    const int n = threadIdx.x;   // 0..255
    h16* Wst = (h16*)(ws + WS_WST);
    h16* Wct = (h16*)(ws + WS_WCT);
    h16* WQt = (h16*)(ws + WS_WQT);
    h16* WKt = (h16*)(ws + WS_WKT);
    h16* WVt = (h16*)(ws + WS_WVT);
    float* bs2 = (float*)(ws + WS_BS2);
    float* bc2 = (float*)(ws + WS_BC2);

    float mc = norm[n], rc = norm[256 + n], mo = norm[512 + n], ro = norm[768 + n];
    float wsv = Ws[(size_t)n * HID_ + c];
    float wcv = Wc[(size_t)n * HID_ + c];
    Wst[(size_t)c * D_ + n] = (h16)(rc * wsv);
    Wct[(size_t)c * D_ + n] = (h16)(ro * wcv);

    red[n] = mc * rc * wsv;
    __syncthreads();
    for (int s = 128; s > 0; s >>= 1) { if (n < s) red[n] += red[n + s]; __syncthreads(); }
    if (n == 0) bs2[c] = bs[c] - red[0];
    __syncthreads();
    red[n] = mo * ro * wcv;
    __syncthreads();
    for (int s = 128; s > 0; s >>= 1) { if (n < s) red[n] += red[n + s]; __syncthreads(); }
    if (n == 0) bc2[c] = bc[c] - red[0];

    const int hh = c >> 6, dd = c & 63;
#pragma unroll
    for (int r = 0; r < 2; ++r) {
        int n2 = n + r * 256;
        size_t src = (size_t)hh * HID_ * HD_ + (size_t)n2 * HD_ + dd;
        size_t dst = (size_t)c * HID_ + n2;
        WQt[dst] = (h16)Wq[src];
        WKt[dst] = (h16)Wk[src];
        WVt[dst] = (h16)Wv[src];
    }
}

// ---------------- fused main kernel ----------------

#define SCR_OFF   262144                 // after ENC
#define WSTG_OFF  (262144 + 36864)       // weight staging region (18432 B)
#define LDS_BYTES (WSTG_OFF + 18432)     // 317440 <= 320KB WGP LDS

__global__ __launch_bounds__(512, 1)
void ca_main(const float* __restrict__ state, const int* __restrict__ aiP,
             const char* __restrict__ ws, float* __restrict__ out) {
    extern __shared__ char smem[];
    h16* ENC = (h16*)smem;                          // [256][512] f16
    char* scratch = smem + SCR_OFF;
    h16*  Xp = (h16*)scratch;                       // phase1: [256][64] f16
    h16*  KV = (h16*)scratch;                       // phase2: [240][64] f16 (K then V)
    float* Qb = (float*)(scratch + 30720);          // [16][64] f32
    float* Sb = (float*)(scratch + 30720 + 4096);   // [240] scores
    float* Wg = (float*)(scratch + 30720 + 4096 + 960); // [240] softmax weights
    h16* Wstg1 = (h16*)(smem + WSTG_OFF);           // phase1: [128][72] f16
    h16* bufA  = (h16*)(smem + WSTG_OFF);           // phase2: [64][72] f16
    h16* bufB  = (h16*)(smem + WSTG_OFF + 9216);    // phase2: [64][72] f16

    const h16* Wst = (const h16*)(ws + WS_WST);
    const h16* Wct = (const h16*)(ws + WS_WCT);
    const h16* WQt = (const h16*)(ws + WS_WQT);
    const h16* WKt = (const h16*)(ws + WS_WKT);
    const h16* WVt = (const h16*)(ws + WS_WVT);
    const float* bs2 = (const float*)(ws + WS_BS2);
    const float* bc2 = (const float*)(ws + WS_BC2);

    const int tid = threadIdx.x;
    const int lane = tid & 31;
    // Wave id as a *provably uniform* scalar: branches on it become s_cbranch
    // (no exec masking around the WMMA loops).
    const int w = __builtin_amdgcn_readfirstlane(tid >> 5);
    const int ai = aiP[0];
    const int btile = blockIdx.x;      // 16 batches per WG

    // X cooperative-load mapping: rows 0..239 = others (b-major, agent-skip ai),
    // rows 240..255 = current-agent rows. 2 threads per row, 32 cols each.
    const int xr = tid >> 1;
    const int xc = (tid & 1) * 32;
    int bl, ag;
    if (xr < OTH_) { bl = xr / 15; int ap = xr % 15; ag = ap + (ap >= ai); }
    else           { bl = xr - OTH_; ag = ai; }
    const float* xsrc = state + (((size_t)(btile * MB_ + bl) * A_ + ag) * D_) + xc;
    h16* xdst = Xp + xr * 64 + xc;

    // phase-1 weight-stage mapping: 128 cols x 64 k per chunk, 32B per thread
    const int p1c = tid >> 2, p1k = (tid & 3) * 16;
    // phase-2 weight-stage mapping: 64 cols x 64 k per chunk, 16B per thread
    const int p2c = tid >> 3, p2k = (tid & 7) * 8;

    // ---- Phase 1: ENC = leaky(X @ W' + b') (BN folded into W', b')
    for (int cp = 0; cp < 4; ++cp) {               // 128 output cols per panel
        v8f acc[8];
#pragma unroll
        for (int t = 0; t < 8; ++t) acc[t] = (v8f){};
        for (int kp = 0; kp < 4; ++kp) {           // stream X + Wct in 64-deep K panels
            __syncthreads();
            {   // async DMA the shared Wct panel into LDS (overlaps X convert below)
                const h16* srcw = Wct + (size_t)(cp * 128 + p1c) * D_ + kp * 64 + p1k;
                h16* dstw = Wstg1 + p1c * 72 + p1k;
                async_cp16(srcw, dstw);
                async_cp16(srcw + 8, dstw + 8);
            }
            {   // X panel f32 -> f16 into LDS
                const float* s = xsrc + kp * 64;
#pragma unroll
                for (int j = 0; j < 32; j += 4) {
                    v4f f = *(const v4f*)(s + j);
                    xdst[j + 0] = (h16)f.x; xdst[j + 1] = (h16)f.y;
                    xdst[j + 2] = (h16)f.z; xdst[j + 3] = (h16)f.w;
                }
            }
            wait_async0();
            __syncthreads();
            if (w < 15) {                          // B from staged LDS (scalar branch)
#pragma unroll
                for (int ks = 0; ks < 2; ++ks) {
                    v16h aF = ldA(Xp, 64, w * 16 + (lane & 15), ks * 32, lane);
#pragma unroll
                    for (int t = 0; t < 8; ++t)
                        acc[t] = WMMA(aF, ldB(Wstg1, t * 16, ks * 32, lane), acc[t]);
                }
            } else {                               // wave 15: s_enc weights from global
#pragma unroll
                for (int ks = 0; ks < 2; ++ks) {
                    v16h aF = ldA(Xp, 64, w * 16 + (lane & 15), ks * 32, lane);
#pragma unroll
                    for (int t = 0; t < 8; ++t)
                        acc[t] = WMMA(aF, gB(Wst, D_, cp * 128 + t * 16,
                                             kp * 64 + ks * 32, lane), acc[t]);
                }
            }
        }
        // epilogue: bias + leaky + f16 store into resident ENC
        const float* b1 = (w == 15) ? bs2 : bc2;
#pragma unroll
        for (int t = 0; t < 8; ++t) {
            int col = cp * 128 + t * 16 + (lane & 15);
            float bb = b1[col];
#pragma unroll
            for (int i = 0; i < 8; ++i) {
                int row = w * 16 + i + ((lane & 16) >> 1);
                float x = acc[t][i] + bb;
                x = (x > 0.f) ? x : 0.3f * x;
                ENC[row * HID_ + col] = (h16)x;
            }
        }
    }
    __syncthreads();

    // ---- Phase 2: per-head K/Q, scores, softmax, V, weighted sum
    const int rtbase = (w < 15) ? w * 16 : OTH_;
    for (int h = 0; h < H_; ++h) {
        {   // K (waves 0..14, weights staged via async DMA) / Q (wave 15, global WQt)
            {
                const h16* s0 = WKt + (size_t)(h * HD_ + p2c) * HID_ + p2k;
                async_cp16(s0, bufA + p2c * 72 + p2k);
            }
            v8f acc[4];
#pragma unroll
            for (int t = 0; t < 4; ++t) acc[t] = (v8f){};
            for (int kc = 0; kc < 8; ++kc) {       // 64-deep chunks, double buffered
                wait_async0();
                __syncthreads();
                if (kc < 7) {
                    const h16* sn = WKt + (size_t)(h * HD_ + p2c) * HID_ + (kc + 1) * 64 + p2k;
                    async_cp16(sn, ((kc & 1) ? bufA : bufB) + p2c * 72 + p2k);
                }
                const h16* bb = (kc & 1) ? bufB : bufA;
                if (w < 15) {                      // scalar branch, unrolled MAC block
#pragma unroll
                    for (int ks = 0; ks < 2; ++ks) {
                        v16h aF = ldA(ENC, HID_, rtbase + (lane & 15),
                                      kc * 64 + ks * 32, lane);
#pragma unroll
                        for (int t = 0; t < 4; ++t)
                            acc[t] = WMMA(aF, ldB(bb, t * 16, ks * 32, lane), acc[t]);
                    }
                } else {
#pragma unroll
                    for (int ks = 0; ks < 2; ++ks) {
                        v16h aF = ldA(ENC, HID_, rtbase + (lane & 15),
                                      kc * 64 + ks * 32, lane);
#pragma unroll
                        for (int t = 0; t < 4; ++t)
                            acc[t] = WMMA(aF, gB(WQt, HID_, h * HD_ + t * 16,
                                                 kc * 64 + ks * 32, lane), acc[t]);
                    }
                }
                __syncthreads();
            }
            if (w < 15) {
#pragma unroll
                for (int t = 0; t < 4; ++t)
#pragma unroll
                    for (int i = 0; i < 8; ++i)
                        KV[(w * 16 + i + ((lane & 16) >> 1)) * HD_ + t * 16 + (lane & 15)] =
                            (h16)acc[t][i];
            } else {
#pragma unroll
                for (int t = 0; t < 4; ++t)
#pragma unroll
                    for (int i = 0; i < 8; ++i)
                        Qb[(i + ((lane & 16) >> 1)) * HD_ + t * 16 + (lane & 15)] = acc[t][i];
            }
        }
        __syncthreads();
        if (tid < OTH_) {                          // scores = q.k / sqrt(hd)
            int b = tid / 15;
            const float* q = Qb + b * HD_;
            const h16* k = KV + tid * HD_;
            float s = 0.f;
#pragma unroll 8
            for (int d = 0; d < HD_; ++d) s += q[d] * (float)k[d];
            Sb[tid] = s * 0.125f;
        }
        __syncthreads();
        if (tid < MB_) {                           // softmax over 15 agents
            float m = -1e30f;
            for (int j = 0; j < 15; ++j) m = fmaxf(m, Sb[tid * 15 + j]);
            float e[15], sum = 0.f;
            for (int j = 0; j < 15; ++j) { e[j] = __expf(Sb[tid * 15 + j] - m); sum += e[j]; }
            float inv = 1.f / sum;
            for (int j = 0; j < 15; ++j) Wg[tid * 15 + j] = e[j] * inv;
        }
        __syncthreads();
        {   // V = leaky(ENC @ WVt), waves 0..14, overwrite KV
            {
                const h16* s0 = WVt + (size_t)(h * HD_ + p2c) * HID_ + p2k;
                async_cp16(s0, bufA + p2c * 72 + p2k);
            }
            v8f acc[4];
#pragma unroll
            for (int t = 0; t < 4; ++t) acc[t] = (v8f){};
            for (int kc = 0; kc < 8; ++kc) {
                wait_async0();
                __syncthreads();
                if (kc < 7) {
                    const h16* sn = WVt + (size_t)(h * HD_ + p2c) * HID_ + (kc + 1) * 64 + p2k;
                    async_cp16(sn, ((kc & 1) ? bufA : bufB) + p2c * 72 + p2k);
                }
                if (w < 15) {
                    const h16* bb = (kc & 1) ? bufB : bufA;
#pragma unroll
                    for (int ks = 0; ks < 2; ++ks) {
                        v16h aF = ldA(ENC, HID_, w * 16 + (lane & 15),
                                      kc * 64 + ks * 32, lane);
#pragma unroll
                        for (int t = 0; t < 4; ++t)
                            acc[t] = WMMA(aF, ldB(bb, t * 16, ks * 32, lane), acc[t]);
                    }
                }
                __syncthreads();
            }
            if (w < 15) {
#pragma unroll
                for (int t = 0; t < 4; ++t)
#pragma unroll
                    for (int i = 0; i < 8; ++i) {
                        float x = acc[t][i];
                        x = (x > 0.f) ? x : 0.3f * x;
                        KV[(w * 16 + i + ((lane & 16) >> 1)) * HD_ + t * 16 + (lane & 15)] =
                            (h16)x;
                    }
            }
        }
        __syncthreads();
#pragma unroll
        for (int r = 0; r < 2; ++r) {              // att[b,d] = sum_a w*v, 2 outputs/thread
            int o = tid + r * 512;
            int b = o >> 6, d = o & 63;
            const float* wg = Wg + b * 15;
            const h16* vv = KV + (b * 15) * HD_ + d;
            float a = 0.f;
#pragma unroll
            for (int ap = 0; ap < 15; ++ap) a += wg[ap] * (float)vv[ap * HD_];
            out[((size_t)(btile * MB_ + b) * HID_) + h * HD_ + d] = a;
        }
        __syncthreads();
    }
}

extern "C" void kernel_launch(void* const* d_in, const int* in_sizes, int n_in,
                              void* d_out, int out_size, void* d_ws, size_t ws_size,
                              hipStream_t stream) {
    const float* state = (const float*)d_in[0];
    const int*   ai    = (const int*)d_in[1];
    const float* Ws    = (const float*)d_in[2];
    const float* bs    = (const float*)d_in[3];
    const float* Wc    = (const float*)d_in[4];
    const float* bc    = (const float*)d_in[5];
    const float* Wk    = (const float*)d_in[6];
    const float* Wq    = (const float*)d_in[7];
    const float* Wv    = (const float*)d_in[8];
    float* out = (float*)d_out;
    char*  ws  = (char*)d_ws;
    float* stats = (float*)(ws + WS_STATS);
    float* norm  = (float*)(ws + WS_NORM);

    (void)in_sizes; (void)n_in; (void)out_size; (void)ws_size;

    hipFuncSetAttribute(reinterpret_cast<const void*>(ca_main),
                        hipFuncAttributeMaxDynamicSharedMemorySize, LDS_BYTES);

    hipLaunchKernelGGL(ca_zero, dim3(4), dim3(256), 0, stream, stats);
    hipLaunchKernelGGL(ca_stats, dim3(512), dim3(256), 0, stream, state, ai, stats);
    hipLaunchKernelGGL(ca_finalize, dim3(1), dim3(256), 0, stream, stats, norm);
    hipLaunchKernelGGL(ca_prep, dim3(512), dim3(256), 0, stream,
                       Ws, bs, Wc, bc, Wk, Wq, Wv, norm, ws);
    hipLaunchKernelGGL(ca_main, dim3(B_ / MB_), dim3(512), LDS_BYTES, stream,
                       state, ai, ws, out);
}
